// SCNN_20375324852603
// MI455X (gfx1250) — matrix operand order repaired and checked by reading, not compiled, not run on hardware
//
#include <hip/hip_runtime.h>

// ---------------- types ----------------
typedef __attribute__((ext_vector_type(16))) __bf16          bf16x16;
typedef __attribute__((ext_vector_type(8)))  float           floatx8;
typedef __attribute__((ext_vector_type(8)))  unsigned short  u16x8;
typedef __attribute__((ext_vector_type(16))) unsigned short  u16x16;

__device__ __forceinline__ unsigned short f2bf(float f) {
  unsigned int u = __builtin_bit_cast(unsigned int, f);
  u += 0x7fffu + ((u >> 16) & 1u);            // round-to-nearest-even
  return (unsigned short)(u >> 16);
}
__device__ __forceinline__ float bf2f(unsigned short s) {
  unsigned int u = ((unsigned int)s) << 16;
  return __builtin_bit_cast(float, u);
}

// ---------------- problem constants ----------------
#define N_IM   8
#define C_IN   128
#define C_MID  1024
#define H_IM   36
#define W_IM   100
#define HW     (H_IM * W_IM)        // 3600
#define NPIX   (N_IM * HW)          // 28800 (multiple of 128)
#define K33    (C_IN * 9)           // 1152
#define K11    C_MID                // 1024
#define LDS_S  40                   // LDS row stride (elems), 80B: conflict-free, 16B aligned
#define BM     64
#define BN     128

// ---------------- f32 -> bf16 convert (weights) ----------------
__global__ void cvt_f32_bf16(const float* __restrict__ s, unsigned short* __restrict__ d, int n) {
  int i = blockIdx.x * blockDim.x + threadIdx.x;
  if (i < n) d[i] = f2bf(s[i]);
}

// ---------------- im2col for dilated 3x3 (dil=4, pad=4) ----------------
// Bp[pix][ci*9 + kh*3 + kw], pixel-major, bf16
__global__ void im2col33(const float* __restrict__ src, unsigned short* __restrict__ Bp) {
  int idx = blockIdx.x * blockDim.x + threadIdx.x;       // pix*128 + ci
  if (idx >= NPIX * C_IN) return;
  int ci = idx & 127;
  int pix = idx >> 7;
  int n = pix / HW, hw = pix - n * HW;
  int h = hw / W_IM, w = hw - h * W_IM;
  const float* s = src + ((size_t)(n * C_IN + ci)) * HW;
  unsigned short* d = Bp + (size_t)pix * K33 + ci * 9;
  #pragma unroll
  for (int kh = 0; kh < 3; ++kh) {
    int h2 = h + 4 * (kh - 1);
    #pragma unroll
    for (int kw = 0; kw < 3; ++kw) {
      int w2 = w + 4 * (kw - 1);
      float v = (h2 >= 0 && h2 < H_IM && w2 >= 0 && w2 < W_IM) ? s[h2 * W_IM + w2] : 0.0f;
      d[kh * 3 + kw] = f2bf(v);
    }
  }
}

// ---------------- pack one MP step's B matrix ----------------
// axisMode 0: scan over rows (H), conv along W. axisMode 1: scan over cols (W), conv along H.
__global__ void mp_pack(const float* __restrict__ z, unsigned short* __restrict__ Bp,
                        int Npos, int posLen, int Npad, int prevIdx, int axisMode) {
  int idx = blockIdx.x * blockDim.x + threadIdx.x;       // nn*128 + ci
  if (idx >= Npad * C_IN) return;
  int ci = idx & 127;
  int nn = idx >> 7;
  unsigned short* d = Bp + (size_t)nn * K33 + ci * 9;
  if (nn >= Npos) {
    #pragma unroll
    for (int t = 0; t < 9; ++t) d[t] = 0;
    return;
  }
  int n = nn / posLen, pos = nn - n * posLen;
  const float* s = z + ((size_t)(n * C_IN + ci)) * HW;
  #pragma unroll
  for (int t = 0; t < 9; ++t) {
    int p2 = pos + t - 4;
    float v = 0.0f;
    if (p2 >= 0 && p2 < posLen)
      v = (axisMode == 0) ? s[prevIdx * W_IM + p2] : s[p2 * W_IM + prevIdx];
    d[t] = f2bf(v);
  }
}

// ---------------- WMMA GEMM: C[M][N] = A[M][K] * B[N][K]^T ----------------
// 256 threads = 8 waves, 64x128 block tile, BK=32, double-buffered LDS (1 barrier / K-step).
// Each wave: 16(M) x 64(N) -> 1 shared A fragment, 4 B fragments, 4 WMMAs per K-step.
// mode 0: out bf16 pixel-major [n][Cout]   (conv1 -> Y1)
// mode 1: out f32 NCHW                     (conv2 -> Z / d_out)
// mode 2: z[addr] += relu(acc)             (message passing update)
__global__ __launch_bounds__(256) void gemm_wmma(
    const unsigned short* __restrict__ A,   // [M][K] bf16
    const unsigned short* __restrict__ B,   // [Npad][K] bf16
    int K, int Nvalid, int mode,
    unsigned short* __restrict__ outBf,     // mode 0
    float* __restrict__ outF,               // mode 1 / 2
    int Cout, int posLen, int curIdx, int axisMode) {
  __shared__ __align__(16) unsigned short As[2][BM * LDS_S];
  __shared__ __align__(16) unsigned short Bs[2][BN * LDS_S];

  int t = threadIdx.x;
  int nBase = blockIdx.x * BN;
  int mBase = blockIdx.y * BM;

  // staging: A 64 rows x 32k -> thread: row t>>2, 8 elems at (t&3)*8
  int arow = t >> 2, ak = (t & 3) * 8;
  // staging: B 128 rows x 32k -> thread: row t>>1, 16 elems at (t&1)*16
  int brow = t >> 1, bk = (t & 1) * 16;

  const unsigned short* gA = A + (size_t)(mBase + arow) * K + ak;
  const unsigned short* gB = B + (size_t)(nBase + brow) * K + bk;

  int wave = t >> 5, lane = t & 31;
  int half = lane >> 4, l15 = lane & 15;
  int mIdx = wave & 3;          // 16-row M tile within 64
  int nIdx = wave >> 2;         // 64-col N slab within 128

  floatx8 zero = {0, 0, 0, 0, 0, 0, 0, 0};
  floatx8 acc[4] = {zero, zero, zero, zero};

  // A fragment: lanes 0-15 -> K {0..7,16..23}; lanes 16-31 -> K {8..15,24..31}
  int aOff = (mIdx * 16 + l15) * LDS_S + half * 8;
  // B fragment (Bs is N-major): lanes 0-15 -> K {0..7,16..23}; lanes 16-31 -> K {8..15,24..31}
  int bOff[4];
  #pragma unroll
  for (int j = 0; j < 4; ++j)
    bOff[j] = (nIdx * 64 + j * 16 + l15) * LDS_S + half * 8;

  int nk = K >> 5;

  // prologue: stage chunk 0 into buffer 0
  *(u16x8*)&As[0][arow * LDS_S + ak]      = *(const u16x8*)gA;
  *(u16x8*)&Bs[0][brow * LDS_S + bk]      = *(const u16x8*)gB;
  *(u16x8*)&Bs[0][brow * LDS_S + bk + 8]  = *(const u16x8*)(gB + 8);
  gA += 32; gB += 32;
  __syncthreads();

  for (int ko = 0; ko < nk; ++ko) {
    int cur = ko & 1, nxt = cur ^ 1;
    if (ko + 1 < nk) {
      *(u16x8*)&As[nxt][arow * LDS_S + ak]     = *(const u16x8*)gA;
      *(u16x8*)&Bs[nxt][brow * LDS_S + bk]     = *(const u16x8*)gB;
      *(u16x8*)&Bs[nxt][brow * LDS_S + bk + 8] = *(const u16x8*)(gB + 8);
      gA += 32; gB += 32;
      __builtin_prefetch(gA, 0, 1);
      __builtin_prefetch(gB, 0, 1);
    }

    const unsigned short* as = As[cur];
    const unsigned short* bs = Bs[cur];

    u16x8 alo = *(const u16x8*)(as + aOff);
    u16x8 ahi = *(const u16x8*)(as + aOff + 16);
    u16x16 au;
    #pragma unroll
    for (int i = 0; i < 8; ++i) { au[i] = alo[i]; au[i + 8] = ahi[i]; }
    bf16x16 af = __builtin_bit_cast(bf16x16, au);

    #pragma unroll
    for (int j = 0; j < 4; ++j) {
      u16x8 bl = *(const u16x8*)(bs + bOff[j]);
      u16x8 bh = *(const u16x8*)(bs + bOff[j] + 16);
      u16x16 bu;
      #pragma unroll
      for (int i = 0; i < 8; ++i) { bu[i] = bl[i]; bu[i + 8] = bh[i]; }
      bf16x16 bfj = __builtin_bit_cast(bf16x16, bu);
      acc[j] = __builtin_amdgcn_wmma_f32_16x16x32_bf16(false, af, false, bfj,
                                                       (short)0, acc[j], false, false);
    }
    __syncthreads();
  }

  // C/D layout: VGPR r, lanes 0-15 -> (M=r, N=lane); lanes 16-31 -> (M=r+8, N=lane-16)
  int co0 = mBase + mIdx * 16 + half * 8;

  #pragma unroll
  for (int j = 0; j < 4; ++j) {
    int nj = nBase + nIdx * 64 + j * 16 + l15;
    if (nj >= Nvalid) continue;
    if (mode == 0) {
      #pragma unroll
      for (int r = 0; r < 8; ++r)
        outBf[(size_t)nj * Cout + co0 + r] = f2bf(acc[j][r]);
    } else if (mode == 1) {
      int n = nj / HW, hw = nj - n * HW;
      #pragma unroll
      for (int r = 0; r < 8; ++r)
        outF[((size_t)(n * Cout + co0 + r)) * HW + hw] = acc[j][r];
    } else {
      int n = nj / posLen, pos = nj - n * posLen;
      size_t base = (size_t)n * C_IN * HW +
                    (axisMode == 0 ? (size_t)(curIdx * W_IM + pos)
                                   : (size_t)(pos * W_IM + curIdx));
      #pragma unroll
      for (int r = 0; r < 8; ++r)
        outF[base + (size_t)(co0 + r) * HW] += fmaxf(acc[j][r], 0.0f);
    }
  }
}

// ---------------- batch-norm stats / normalize ----------------
__global__ void stats_pm(const unsigned short* __restrict__ y, int C, int Np,
                         float* __restrict__ mean, float* __restrict__ rstd) {
  int c = blockIdx.x;
  float s = 0.f, s2 = 0.f;
  for (int p = threadIdx.x; p < Np; p += 256) {
    float v = bf2f(y[(size_t)p * C + c]);
    s += v; s2 += v * v;
  }
  __shared__ float sh[256], sh2[256];
  sh[threadIdx.x] = s; sh2[threadIdx.x] = s2;
  __syncthreads();
  for (int o = 128; o > 0; o >>= 1) {
    if ((int)threadIdx.x < o) {
      sh[threadIdx.x] += sh[threadIdx.x + o];
      sh2[threadIdx.x] += sh2[threadIdx.x + o];
    }
    __syncthreads();
  }
  if (threadIdx.x == 0) {
    float m = sh[0] / (float)Np;
    float var = sh2[0] / (float)Np - m * m;
    mean[c] = m; rstd[c] = rsqrtf(var + 1e-5f);
  }
}

__global__ void norm_relu_pm(unsigned short* __restrict__ y, int C, long total,
                             const float* __restrict__ mean, const float* __restrict__ rstd) {
  long i = (long)blockIdx.x * blockDim.x + threadIdx.x;
  if (i >= total) return;
  int c = (int)(i % C);
  float v = (bf2f(y[i]) - mean[c]) * rstd[c];
  y[i] = f2bf(fmaxf(v, 0.f));
}

__global__ void stats_nchw(const float* __restrict__ z, int C,
                           float* __restrict__ mean, float* __restrict__ rstd) {
  int c = blockIdx.x;
  float s = 0.f, s2 = 0.f;
  for (int i = threadIdx.x; i < N_IM * HW; i += 256) {
    int n = i / HW, hw = i - n * HW;
    float v = z[((size_t)(n * C + c)) * HW + hw];
    s += v; s2 += v * v;
  }
  __shared__ float sh[256], sh2[256];
  sh[threadIdx.x] = s; sh2[threadIdx.x] = s2;
  __syncthreads();
  for (int o = 128; o > 0; o >>= 1) {
    if ((int)threadIdx.x < o) {
      sh[threadIdx.x] += sh[threadIdx.x + o];
      sh2[threadIdx.x] += sh2[threadIdx.x + o];
    }
    __syncthreads();
  }
  if (threadIdx.x == 0) {
    float m = sh[0] / (float)(N_IM * HW);
    float var = sh2[0] / (float)(N_IM * HW) - m * m;
    mean[c] = m; rstd[c] = rsqrtf(var + 1e-5f);
  }
}

__global__ void norm_relu_nchw(float* __restrict__ z, int C, int total,
                               const float* __restrict__ mean, const float* __restrict__ rstd) {
  int i = blockIdx.x * blockDim.x + threadIdx.x;
  if (i >= total) return;
  int c = (i / HW) % C;
  z[i] = fmaxf((z[i] - mean[c]) * rstd[c], 0.f);
}

// ---------------- host ----------------
extern "C" void kernel_launch(void* const* d_in, const int* in_sizes, int n_in,
                              void* d_out, int out_size, void* d_ws, size_t ws_size,
                              hipStream_t stream) {
  (void)in_sizes; (void)n_in; (void)out_size; (void)ws_size;
  const float* x   = (const float*)d_in[0];
  const float* w1  = (const float*)d_in[1];
  const float* w2  = (const float*)d_in[2];
  const float* wmp[4] = { (const float*)d_in[3], (const float*)d_in[4],
                          (const float*)d_in[5], (const float*)d_in[6] };
  float* out = (float*)d_out;

  char* ws = (char*)d_ws;
  size_t off = 0;
  auto take = [&](size_t bytes) -> char* {
    off = (off + 255) & ~(size_t)255;
    char* p = ws + off;
    off += bytes;
    return p;
  };

  unsigned short* A1 = (unsigned short*)take((size_t)C_MID * K33 * 2);
  unsigned short* A2 = (unsigned short*)take((size_t)C_IN * K11 * 2);
  unsigned short* Amp[4];
  for (int i = 0; i < 4; ++i) Amp[i] = (unsigned short*)take((size_t)C_IN * K33 * 2);
  unsigned short* Bp  = (unsigned short*)take((size_t)NPIX * K33 * 2);
  unsigned short* Y1  = (unsigned short*)take((size_t)NPIX * C_MID * 2);
  float*          Z   = (float*)take((size_t)NPIX * C_IN * 4);
  unsigned short* Bmp = (unsigned short*)take((size_t)896 * K33 * 2);
  float* Mean = (float*)take(1024 * 4);
  float* Rstd = (float*)take(1024 * 4);

  auto cvt = [&](const float* s, unsigned short* d, int n) {
    cvt_f32_bf16<<<(n + 255) / 256, 256, 0, stream>>>(s, d, n);
  };
  cvt(w1, A1, C_MID * K33);
  cvt(w2, A2, C_IN * K11);
  for (int i = 0; i < 4; ++i) cvt(wmp[i], Amp[i], C_IN * K33);

  auto layer1 = [&](const float* src, float* dst) {
    im2col33<<<(NPIX * C_IN + 255) / 256, 256, 0, stream>>>(src, Bp);
    gemm_wmma<<<dim3(NPIX / BN, C_MID / BM), 256, 0, stream>>>(
        A1, Bp, K33, NPIX, 0, Y1, nullptr, C_MID, 0, 0, 0);
    stats_pm<<<C_MID, 256, 0, stream>>>(Y1, C_MID, NPIX, Mean, Rstd);
    long tot1 = (long)NPIX * C_MID;
    norm_relu_pm<<<(unsigned)((tot1 + 255) / 256), 256, 0, stream>>>(Y1, C_MID, tot1, Mean, Rstd);
    gemm_wmma<<<dim3(NPIX / BN, C_IN / BM), 256, 0, stream>>>(
        A2, Y1, K11, NPIX, 1, nullptr, dst, C_IN, 0, 0, 0);
    stats_nchw<<<C_IN, 256, 0, stream>>>(dst, C_IN, Mean, Rstd);
    int tot2 = NPIX * C_IN;
    norm_relu_nchw<<<(tot2 + 255) / 256, 256, 0, stream>>>(dst, C_IN, tot2, Mean, Rstd);
  };

  layer1(x, Z);

  auto mp = [&](const unsigned short* Aw, int axisMode, bool rev) {
    int posLen = (axisMode == 0) ? W_IM : H_IM;   // conv length
    int scanLen = (axisMode == 0) ? H_IM : W_IM;  // scan steps
    int Npos = N_IM * posLen;                     // 800 or 288
    int Npad = (Npos + BN - 1) & ~(BN - 1);       // 896 or 384
    for (int s = 1; s < scanLen; ++s) {
      int cur  = rev ? (scanLen - 1 - s) : s;
      int prev = rev ? (cur + 1) : (cur - 1);
      mp_pack<<<(Npad * C_IN + 255) / 256, 256, 0, stream>>>(
          Z, Bmp, Npos, posLen, Npad, prev, axisMode);
      gemm_wmma<<<dim3(Npad / BN, C_IN / BM), 256, 0, stream>>>(
          Aw, Bmp, K33, Npos, 2, nullptr, Z, C_IN, posLen, cur, axisMode);
    }
  };
  mp(Amp[0], 0, false);   // UD
  mp(Amp[1], 0, true);    // DU
  mp(Amp[2], 1, false);   // LR
  mp(Amp[3], 1, true);    // RL

  layer1(Z, out);
}
